// Head_14242111553650
// MI455X (gfx1250) — matrix operand order repaired and checked by reading, not compiled, not run on hardware
//
#include <hip/hip_runtime.h>

#define Bb 8
#define Tt 2048
#define Cc 1024
#define Hh 64

typedef __attribute__((ext_vector_type(16))) __bf16 v16bf;
typedef __attribute__((ext_vector_type(2)))  __bf16 v2bf;
typedef __attribute__((ext_vector_type(8)))  float  v8f;

struct U4 { unsigned int x, y, z, w; };
struct F4 { float x, y, z, w; };
union  V16 { v16bf v; U4 u[2]; v2bf p[8]; };

#if defined(__has_builtin)
#  if __has_builtin(__builtin_amdgcn_cvt_pk_bf16_f32)
#    define HAVE_PK_BF16 1
#  endif
#endif

__device__ __forceinline__ unsigned short f2bfraw(float f) {
  union { float f; unsigned u; } c; c.f = f;
  unsigned r = c.u + 0x7FFFu + ((c.u >> 16) & 1u);   // round-to-nearest-even
  return (unsigned short)(r >> 16);
}
__device__ __forceinline__ __bf16 f2bf(float f) {
  union { unsigned short s; __bf16 b; } o; o.s = f2bfraw(f);
  return o.b;
}
__device__ __forceinline__ v2bf pkbf(float a, float b) {
#if defined(HAVE_PK_BF16)
  return __builtin_amdgcn_cvt_pk_bf16_f32(a, b);     // v_cvt_pk_bf16_f32
#else
  v2bf r; r[0] = f2bf(a); r[1] = f2bf(b); return r;
#endif
}

// Async global->LDS copy, 16 bytes per lane (ASYNCcnt-tracked).
// Generic pointers to __shared__ carry the LDS offset in addr[31:0] (ISA 10.2).
__device__ __forceinline__ void async_copy16(unsigned short* lds_dst,
                                             const unsigned short* gsrc) {
  unsigned loff = (unsigned)(size_t)(void*)lds_dst;
  asm volatile("global_load_async_to_lds_b128 %0, %1, off"
               :: "v"(loff), "v"(gsrc) : "memory");
}
__device__ __forceinline__ void wait_async() {
  asm volatile("s_wait_asynccnt 0" ::: "memory");
}

// ---------------------------------------------------------------------------
// Kernel 0: W [C,H] fp32 -> Wt [H,C] bf16 (WMMA B-operand contiguous in K)
// ---------------------------------------------------------------------------
__global__ __launch_bounds__(256) void wt_kernel(const float* __restrict__ Wq,
                                                 const float* __restrict__ Wk,
                                                 const float* __restrict__ Wv,
                                                 unsigned short* __restrict__ WtAll) {
  int idx = blockIdx.x * 256 + threadIdx.x;          // 0 .. C*H-1
  int mat = blockIdx.y;
  const float* W = (mat == 0) ? Wq : (mat == 1) ? Wk : Wv;
  int c = idx >> 6;
  int n = idx & 63;
  WtAll[(size_t)mat * (Hh * Cc) + (size_t)n * Cc + c] = f2bfraw(W[idx]);
}

// ---------------------------------------------------------------------------
// Kernel 1: fused QKV projection. 8 waves/block, one wave = 16 rows x 64 cols
// of q, k AND vT. The Wt K-chunk (12KB, identical for all 8 waves) is staged
// into LDS with async copies, double-buffered; B operands then come from LDS
// at ds latency while the async engine fetches the next chunk.
// LDS chunk layout: [m][col][kk] = WLds[m*2048 + col*32 + kk], kk = K-in-chunk.
// ---------------------------------------------------------------------------
__global__ __launch_bounds__(256) void qkv_kernel(const float* __restrict__ x,
                                                  const unsigned short* __restrict__ WtAll,
                                                  unsigned short* __restrict__ qo,
                                                  unsigned short* __restrict__ ko,
                                                  unsigned short* __restrict__ vto) {
  __shared__ unsigned short WLds[2][3 * 64 * 32];  // 2 x 12KB

  const int tid  = threadIdx.x;
  const int lane = tid & 31;
  const int wave = tid >> 5;
  const int half = lane >> 4;
  const int l16  = lane & 15;
  const int m0   = (blockIdx.x * 8 + wave) * 16;

  const int row = m0 + l16;                      // A-layout: M = lane mod 16
  const float* xr = x + (size_t)row * Cc;

  v8f acc[12];
  #pragma unroll
  for (int i = 0; i < 12; ++i)
    #pragma unroll
    for (int r = 0; r < 8; ++r) acc[i][r] = 0.0f;

  // stage one 32-wide K-chunk of all three Wt matrices (3 x b128 per thread)
  auto stageW = [&](int kc, int buf) {
    #pragma unroll
    for (int i = 0; i < 3; ++i) {
      int flat = (i * 256 + tid) * 8;            // 0..6136
      int m    = flat >> 11;
      int rem  = flat & 2047;
      int col  = rem >> 5;
      int kk   = rem & 31;
      async_copy16(&WLds[buf][flat],
                   WtAll + (size_t)m * (Hh * Cc) + (size_t)col * Cc + kc + kk);
    }
  };

  stageW(0, 0);
  wait_async();
  __syncthreads();

  for (int kc = 0; kc < Cc; kc += 32) {
    const int buf = (kc >> 5) & 1;
    if (kc + 32 < Cc) stageW(kc + 32, buf ^ 1);  // prefetch next Wt chunk

    // ---- A operand from global x: K(e)=kc+half*8+(e<8?e:e+8) ----
    const int a1 = kc + half * 8;
    F4 xa = *(const F4*)(xr + a1);
    F4 xb = *(const F4*)(xr + a1 + 4);
    F4 xc = *(const F4*)(xr + a1 + 16);
    F4 xd = *(const F4*)(xr + a1 + 20);
    V16 a;
    a.p[0] = pkbf(xa.x, xa.y);  a.p[1] = pkbf(xa.z, xa.w);
    a.p[2] = pkbf(xb.x, xb.y);  a.p[3] = pkbf(xb.z, xb.w);
    a.p[4] = pkbf(xc.x, xc.y);  a.p[5] = pkbf(xc.z, xc.w);
    a.p[6] = pkbf(xd.x, xd.y);  a.p[7] = pkbf(xd.z, xd.w);

    // ---- B operands from LDS, 12 WMMAs ----
    #pragma unroll
    for (int m = 0; m < 3; ++m)
      #pragma unroll
      for (int nt = 0; nt < 4; ++nt) {
        const unsigned short* wp =
            &WLds[buf][m * 2048 + (nt * 16 + l16) * 32 + half * 8];
        V16 b;
        b.u[0] = *(const U4*)(wp);
        b.u[1] = *(const U4*)(wp + 16);
        acc[m * 4 + nt] = __builtin_amdgcn_wmma_f32_16x16x32_bf16(
            false, a.v, false, b.v, (short)0, acc[m * 4 + nt], false, false);
      }

    wait_async();            // next Wt chunk landed
    __syncthreads();         // all waves done reading buf
  }

  // q, k row-major; v stored transposed as vT[b][h][t]
  #pragma unroll
  for (int nt = 0; nt < 4; ++nt)
    #pragma unroll
    for (int r = 0; r < 8; ++r) {
      int mm = m0 + r + half * 8;              // C/D layout: M = r + 8*half
      int nn = nt * 16 + l16;
      qo[(size_t)mm * Hh + nn] = f2bfraw(acc[nt][r]);
      ko[(size_t)mm * Hh + nn] = f2bfraw(acc[4 + nt][r]);
      int bi = mm >> 11, tt = mm & (Tt - 1);
      vto[((size_t)bi * Hh + nn) * Tt + tt] = f2bfraw(acc[8 + nt][r]);
    }
}

// ---------------------------------------------------------------------------
// Kernel 2: flash attention. 8 waves/block share K/V tiles staged into LDS
// with async global->LDS copies, double-buffered. Uniform trip count so
// __syncthreads is legal; waves past their causal range skip compute only.
// ---------------------------------------------------------------------------
__global__ __launch_bounds__(256) void attn_kernel(const unsigned short* __restrict__ q,
                                                   const unsigned short* __restrict__ k,
                                                   const unsigned short* __restrict__ vT,
                                                   float* __restrict__ out) {
  __shared__ unsigned short Ktile[2][64 * 64];     // [buf][s_local*64 + h]   8KB each
  __shared__ unsigned short Vtile[2][64 * 64];     // [buf][h*64 + s_local]   8KB each
  __shared__ unsigned short Plds[8][16 * 64];      // per-wave P staging     16KB

  const int tid  = threadIdx.x;
  const int lane = tid & 31;
  const int wave = tid >> 5;
  const int half = lane >> 4;
  const int l16  = lane & 15;
  const int b    = blockIdx.y;
  const int m0   = blockIdx.x * 128 + wave * 16;

  const unsigned short* qb = q  + (size_t)b * Tt * Hh;
  const unsigned short* kb = k  + (size_t)b * Tt * Hh;
  const unsigned short* vb = vT + (size_t)b * Hh * Tt;

  // Q in A-layout (H=64 -> 2 K-chunks of 32)
  v16bf qa[2];
  {
    const int qrow = m0 + l16;
    #pragma unroll
    for (int cc = 0; cc < 2; ++cc) {
      const unsigned short* qp = qb + (size_t)qrow * Hh + cc * 32 + half * 8;
      V16 t;
      t.u[0] = *(const U4*)(qp);
      t.u[1] = *(const U4*)(qp + 16);
      qa[cc] = t.v;
    }
  }

  float m_i[8], l_i[8];
  v8f o[4];
  #pragma unroll
  for (int r = 0; r < 8; ++r) { m_i[r] = -1e30f; l_i[r] = 0.0f; }
  #pragma unroll
  for (int nt = 0; nt < 4; ++nt)
    #pragma unroll
    for (int r = 0; r < 8; ++r) o[nt][r] = 0.0f;

  const int   jlastB = blockIdx.x * 2 + 1;         // uniform across block
  const float scale  = 0.03125f;                   // 1/sqrt(C)

  // issue async stage of key-block jb into buffer buf (4 x b128 per thread)
  auto stage = [&](int jb, int buf) {
    const int s0 = jb * 64;
    const unsigned short* kg = kb + (size_t)s0 * Hh;      // 4096 contiguous bf16
    #pragma unroll
    for (int i = 0; i < 2; ++i) {
      int idx = (i * 256 + tid) * 8;                      // 0..4088
      async_copy16(&Ktile[buf][idx], kg + idx);
      int hrow = idx >> 6, sc = idx & 63;
      async_copy16(&Vtile[buf][idx], vb + (size_t)hrow * Tt + s0 + sc);
    }
  };

  stage(0, 0);
  wait_async();
  __syncthreads();

  for (int jb = 0; jb <= jlastB; ++jb) {
    const int buf = jb & 1;
    if (jb < jlastB) stage(jb + 1, buf ^ 1);       // prefetch next tile
    const int s0 = jb * 64;

    if (s0 <= m0 + 15) {                           // causal range of this wave
      // ---- S = Q * K^T ----
      V16 bk[8];
      #pragma unroll
      for (int st = 0; st < 4; ++st)
        #pragma unroll
        for (int cc = 0; cc < 2; ++cc) {
          const unsigned short* kp =
              &Ktile[buf][(st * 16 + l16) * 64 + cc * 32 + half * 8];
          bk[st * 2 + cc].u[0] = *(const U4*)(kp);
          bk[st * 2 + cc].u[1] = *(const U4*)(kp + 16);
        }
      v8f sv[4];
      #pragma unroll
      for (int st = 0; st < 4; ++st) {
        v8f c;
        #pragma unroll
        for (int r = 0; r < 8; ++r) c[r] = 0.0f;
        c = __builtin_amdgcn_wmma_f32_16x16x32_bf16(
            false, qa[0], false, bk[st * 2 + 0].v, (short)0, c, false, false);
        c = __builtin_amdgcn_wmma_f32_16x16x32_bf16(
            false, qa[1], false, bk[st * 2 + 1].v, (short)0, c, false, false);
        #pragma unroll
        for (int r = 0; r < 8; ++r) sv[st][r] = c[r] * scale;
      }

      // ---- causal mask on diagonal blocks ----
      if (s0 + 63 > m0) {
        #pragma unroll
        for (int st = 0; st < 4; ++st)
          #pragma unroll
          for (int r = 0; r < 8; ++r) {
            int tq = m0 + r + half * 8;
            int sk = s0 + st * 16 + l16;
            if (sk > tq) sv[st][r] = -1e30f;
          }
      }

      // ---- online softmax ----
      float alpha[8];
      #pragma unroll
      for (int r = 0; r < 8; ++r) {
        float v = fmaxf(fmaxf(sv[0][r], sv[1][r]), fmaxf(sv[2][r], sv[3][r]));
        #pragma unroll
        for (int d = 1; d < 16; d <<= 1) v = fmaxf(v, __shfl_xor(v, d, 32));
        float mn = fmaxf(m_i[r], v);
        alpha[r] = __expf(m_i[r] - mn);
        m_i[r] = mn;
      }
      #pragma unroll
      for (int st = 0; st < 4; ++st)
        #pragma unroll
        for (int r = 0; r < 8; ++r) sv[st][r] = __expf(sv[st][r] - m_i[r]);
      #pragma unroll
      for (int r = 0; r < 8; ++r) {
        float v = sv[0][r] + sv[1][r] + sv[2][r] + sv[3][r];
        #pragma unroll
        for (int d = 1; d < 16; d <<= 1) v += __shfl_xor(v, d, 32);
        l_i[r] = l_i[r] * alpha[r] + v;
      }
      #pragma unroll
      for (int nt = 0; nt < 4; ++nt)
        #pragma unroll
        for (int r = 0; r < 8; ++r) o[nt][r] *= alpha[r];

      // ---- relayout P: C-layout -> A-layout via per-wave LDS tile ----
      unsigned short* pl = &Plds[wave][0];
      #pragma unroll
      for (int st = 0; st < 4; ++st)
        #pragma unroll
        for (int r = 0; r < 8; ++r)
          pl[(r + half * 8) * 64 + st * 16 + l16] = f2bfraw(sv[st][r]);
      asm volatile("s_wait_dscnt 0" ::: "memory");

      v16bf pa[2];
      #pragma unroll
      for (int cc = 0; cc < 2; ++cc) {
        const unsigned short* pp = pl + l16 * 64 + cc * 32 + half * 8;
        V16 t;
        t.u[0] = *(const U4*)(pp);
        t.u[1] = *(const U4*)(pp + 16);
        pa[cc] = t.v;
      }

      // ---- O += P * V ----
      V16 bv[8];
      #pragma unroll
      for (int nt = 0; nt < 4; ++nt)
        #pragma unroll
        for (int cc = 0; cc < 2; ++cc) {
          const unsigned short* vp =
              &Vtile[buf][(nt * 16 + l16) * 64 + cc * 32 + half * 8];
          bv[nt * 2 + cc].u[0] = *(const U4*)(vp);
          bv[nt * 2 + cc].u[1] = *(const U4*)(vp + 16);
        }
      #pragma unroll
      for (int nt = 0; nt < 4; ++nt) {
        o[nt] = __builtin_amdgcn_wmma_f32_16x16x32_bf16(
            false, pa[0], false, bv[nt * 2 + 0].v, (short)0, o[nt], false, false);
        o[nt] = __builtin_amdgcn_wmma_f32_16x16x32_bf16(
            false, pa[1], false, bv[nt * 2 + 1].v, (short)0, o[nt], false, false);
      }
    }

    wait_async();            // staged tile for jb+1 landed
    __syncthreads();         // everyone done reading buf, writes visible
  }

  // ---- epilogue: out = O / l ----
  float* ob = out + (size_t)b * Tt * Hh;
  #pragma unroll
  for (int nt = 0; nt < 4; ++nt)
    #pragma unroll
    for (int r = 0; r < 8; ++r) {
      int tq = m0 + r + half * 8;
      ob[(size_t)tq * Hh + nt * 16 + l16] = o[nt][r] / l_i[r];
    }
}

// ---------------------------------------------------------------------------
extern "C" void kernel_launch(void* const* d_in, const int* in_sizes, int n_in,
                              void* d_out, int out_size, void* d_ws, size_t ws_size,
                              hipStream_t stream) {
  (void)in_sizes; (void)n_in; (void)out_size; (void)ws_size;
  const float* x  = (const float*)d_in[0];
  const float* Wq = (const float*)d_in[1];
  const float* Wk = (const float*)d_in[2];
  const float* Wv = (const float*)d_in[3];
  float* out = (float*)d_out;

  unsigned short* ws = (unsigned short*)d_ws;
  unsigned short* Wt = ws;                               // 3 * 64 * 1024
  unsigned short* qo = Wt + 3 * Hh * Cc;                 // B*T*H bf16
  unsigned short* ko = qo + (size_t)Bb * Tt * Hh;
  unsigned short* vt = ko + (size_t)Bb * Tt * Hh;        // vT: [B][H][T]

  wt_kernel <<<dim3((Cc * Hh) / 256, 3), 256, 0, stream>>>(Wq, Wk, Wv, Wt);
  qkv_kernel<<<dim3((Bb * Tt) / 128), 256, 0, stream>>>(x, Wt, qo, ko, vt);
  attn_kernel<<<dim3(Tt / 128, Bb), 256, 0, stream>>>(qo, ko, vt, out);
}
